// DynamicMemory_56066503082547
// MI455X (gfx1250) — matrix operand order repaired and checked by reading, not compiled, not run on hardware
//
#include <hip/hip_runtime.h>
#include <hip/hip_bf16.h>

// ---------------- problem constants (fixed by the reference) ----------------
#define E_DIM 256
#define N_DIM 4096
#define B_DIM 64
#define ROWS  32      // n-rows per workgroup tile
#define LPAD  264     // padded LDS row stride in bf16 elements (breaks bank conflicts)

typedef __attribute__((ext_vector_type(8)))  float        v8f;
typedef __attribute__((ext_vector_type(16))) __bf16       v16bf;
typedef __attribute__((ext_vector_type(4)))  unsigned int u32x4;
typedef __attribute__((ext_vector_type(4)))  int          i32x4;
typedef __attribute__((ext_vector_type(8)))  int          i32x8;
typedef __attribute__((address_space(3))) unsigned short  lds_us;

#ifndef __has_builtin
#define __has_builtin(x) 0
#endif
#if __has_builtin(__builtin_amdgcn_tensor_load_to_lds)
#define HAVE_TDM 1
#endif
#if __has_builtin(__builtin_amdgcn_global_load_async_to_lds_b128)
#define HAVE_ASYNC_LDS 1
typedef __attribute__((address_space(1))) i32x4 gi32x4;   // global int4
typedef __attribute__((address_space(3))) i32x4 si32x4;   // LDS int4
#endif

static __device__ __forceinline__ unsigned short f32_to_bf16_rne(float f) {
    unsigned int u = __float_as_uint(f);
    u += 0x7FFFu + ((u >> 16) & 1u);      // round to nearest even
    return (unsigned short)(u >> 16);
}

// ---------------- prep kernels (tiny vs. the fused pass) --------------------

// Ut[e*E + f] = bf16(U[f*E + e])  -- B-operand friendly [K][N] layout
__global__ void prep_transpose_U(const float* __restrict__ U,
                                 unsigned short* __restrict__ Ut) {
    const int e = blockIdx.x;
    const int f = threadIdx.x;
    Ut[e * E_DIM + f] = f32_to_bf16_rne(U[f * E_DIM + e]);
}

// wv[b,f] = sum_e w_emb[b,e] * V[f,e]
__global__ void prep_wv(const float* __restrict__ w_emb,
                        const float* __restrict__ V,
                        float* __restrict__ wvv) {
    const int b = blockIdx.x, f = threadIdx.x;
    __shared__ float wr[E_DIM];
    wr[f] = w_emb[b * E_DIM + f];
    __syncthreads();
    float s = 0.f;
    #pragma unroll 8
    for (int e = 0; e < E_DIM; ++e) s += wr[e] * V[f * E_DIM + e];
    wvv[b * E_DIM + f] = s;
}

// xw[n,f] = sum_e x[n,e] * W[f,e]
__global__ void prep_xw(const float* __restrict__ x,
                        const float* __restrict__ W,
                        float* __restrict__ xw) {
    const int n = blockIdx.x, f = threadIdx.x;
    __shared__ float xr[E_DIM];
    xr[f] = x[(size_t)n * E_DIM + f];
    __syncthreads();
    float s = 0.f;
    #pragma unroll 8
    for (int e = 0; e < E_DIM; ++e) s += xr[e] * W[f * E_DIM + e];
    xw[(size_t)n * E_DIM + f] = s;
}

// xd[b,n] = sum_e x[n,e] * w_emb[b,e]
__global__ void prep_xdot(const float* __restrict__ x,
                          const float* __restrict__ w_emb,
                          float* __restrict__ xd) {
    const int b = blockIdx.y;
    const int n = blockIdx.x * 256 + threadIdx.x;
    __shared__ float wr[E_DIM];
    wr[threadIdx.x] = w_emb[b * E_DIM + threadIdx.x];
    __syncthreads();
    const float* xr = x + (size_t)n * E_DIM;
    float s = 0.f;
    #pragma unroll 8
    for (int e = 0; e < E_DIM; ++e) s += xr[e] * wr[e];
    xd[(size_t)b * N_DIM + n] = s;
}

// ---------------- fused main kernel -----------------------------------------
// grid: B_DIM * (N_DIM/ROWS) blocks, 512 threads (16 waves of 32)
__launch_bounds__(512, 1)
__global__ void dm_fused(const float* __restrict__ x,
                         const float* __restrict__ h,
                         const float* __restrict__ bias,
                         const float* __restrict__ alpha,
                         const unsigned short* __restrict__ Ut,   // bf16 bits [e][f]
                         const float* __restrict__ xw,
                         const float* __restrict__ wvv,
                         const float* __restrict__ xdot,
                         float* __restrict__ out) {
    extern __shared__ char smem_raw[];
    unsigned short* sU = reinterpret_cast<unsigned short*>(smem_raw); // [E_DIM][LPAD]
    unsigned short* sH = sU + E_DIM * LPAD;                           // [ROWS][LPAD]
    float* gate   = reinterpret_cast<float*>(sH + ROWS * LPAD);       // [ROWS]
    float* gdot   = gate + ROWS;                                      // [ROWS]
    float* rowsum = gdot + ROWS;                                      // [ROWS]

    const int t    = threadIdx.x;
    const int lane = t & 31;
    const int wid  = t >> 5;
    const int b    = blockIdx.x >> 7;            // / (N_DIM/ROWS)
    const int n0   = (blockIdx.x & 127) * ROWS;

    if (t < ROWS) rowsum[t] = 0.0f;

    // ---- stage U^T (bf16) into LDS (padded rows) ----
#ifdef HAVE_TDM
    // One TDM descriptor moves the whole 256x256 bf16 tile; the DMA engine
    // inserts the 16B row padding (pad_interval=6 -> every 128 DWORDs,
    // pad_amount=3 -> 4 DWORDs) that de-conflicts the LDS banks.
    if (wid == 0) {
        const unsigned long long ga = (unsigned long long)(const void*)Ut;
        const unsigned ldsoff = (unsigned)(unsigned long long)(lds_us*)sU;
        u32x4 g0;
        g0[0] = 1u;                                   // count=1 (valid), no gather
        g0[1] = ldsoff;                               // lds_addr (bytes)
        g0[2] = (unsigned)(ga & 0xffffffffull);       // global_addr[31:0]
        g0[3] = (unsigned)((ga >> 32) & 0x01ffffffull) | (2u << 30); // addr[56:32] | type=2
        i32x8 g1;
        g1[0] = (int)((1u << 16)      // data_size = 2 bytes
                    | (1u << 20)      // pad_enable
                    | (6u << 22)      // pad_interval: 128 DWORDs (512B rows)
                    | (3u << 25));    // pad_amount: 4 DWORDs (16B)
        g1[1] = (int)(256u << 16);    // tensor_dim0 lo16 (=256 elems)
        g1[2] = (int)(256u << 16);    // tensor_dim0 hi16=0 | tensor_dim1 lo16 (=256)
        g1[3] = (int)(256u << 16);    // tensor_dim1 hi16=0 | tile_dim0 (=256)
        g1[4] = 256;                  // tile_dim1=256 | tile_dim2=0
        g1[5] = 256;                  // tensor_dim0_stride lo32 (=256 elems)
        g1[6] = 0;                    // stride hi | tensor_dim1_stride lo
        g1[7] = 0;
        i32x4 gz4 = {0, 0, 0, 0};
        i32x8 gz8 = {0, 0, 0, 0, 0, 0, 0, 0};
        __builtin_amdgcn_tensor_load_to_lds(g0, g1, gz4, gz4, gz8, 0);
#if __has_builtin(__builtin_amdgcn_s_wait_tensorcnt)
        __builtin_amdgcn_s_wait_tensorcnt(0);
#else
        asm volatile("s_wait_tensorcnt 0" ::: "memory");
#endif
    }
#else
    {
        const int nChunks = E_DIM * (E_DIM / 8);  // 8192 chunks of 8 bf16
        for (int c = t; c < nChunks; c += 512) {
            const int e = c >> 5;
            const int p = (c & 31) * 8;
#ifdef HAVE_ASYNC_LDS
            __builtin_amdgcn_global_load_async_to_lds_b128(
                (gi32x4*)(Ut + (size_t)e * E_DIM + p),
                (si32x4*)(sU + e * LPAD + p), 0, 0);
#else
            u32x4 v = *reinterpret_cast<const u32x4*>(Ut + (size_t)e * E_DIM + p);
            *reinterpret_cast<u32x4*>(sU + e * LPAD + p) = v;
#endif
        }
    }
#endif

    // ---- stage h tile f32 -> bf16 LDS; fused gate dot with x ----
    {
        const int r  = t >> 4;            // 0..31 : row in tile
        const int cb = (t & 15) * 16;     // 16 consecutive cols per thread
        const float4* hp = reinterpret_cast<const float4*>(
            h + ((size_t)b * N_DIM + (n0 + r)) * E_DIM + cb);
        const float4* xp = reinterpret_cast<const float4*>(
            x + (size_t)(n0 + r) * E_DIM + cb);
        float part = 0.0f;
        union { unsigned short s[16]; u32x4 q[2]; } pk;
        #pragma unroll
        for (int i = 0; i < 4; ++i) {
            float4 hv = hp[i];
            float4 xv = xp[i];
            part += hv.x * xv.x + hv.y * xv.y + hv.z * xv.z + hv.w * xv.w;
            pk.s[4 * i + 0] = f32_to_bf16_rne(hv.x);
            pk.s[4 * i + 1] = f32_to_bf16_rne(hv.y);
            pk.s[4 * i + 2] = f32_to_bf16_rne(hv.z);
            pk.s[4 * i + 3] = f32_to_bf16_rne(hv.w);
        }
        unsigned short* dst = sH + r * LPAD + cb;
        *reinterpret_cast<u32x4*>(dst)     = pk.q[0];
        *reinterpret_cast<u32x4*>(dst + 8) = pk.q[1];
        #pragma unroll
        for (int m = 8; m >= 1; m >>= 1) part += __shfl_xor(part, m, 16);
        if ((t & 15) == 0) gdot[r] = part;
    }

#if !defined(HAVE_TDM) && defined(HAVE_ASYNC_LDS)
    asm volatile("s_wait_asynccnt 0" ::: "memory");
#endif
    __syncthreads();

    if (t < ROWS) {
        const float z = gdot[t] + xdot[(size_t)b * N_DIM + n0 + t];
        gate[t] = 1.0f / (1.0f + __expf(-z));
    }
    __syncthreads();

    // ---- WMMA: pre[16x256] = h_bf16[16x256] @ Ut_bf16 ----
    const int rh   = wid >> 3;      // row-half of the 32-row tile
    const int fb0  = wid & 7;       // this wave's two f-blocks
    const int fb1  = fb0 + 8;
    const int half = lane >> 4;
    const int lrow = lane & 15;

    v8f acc0 = {0.f, 0.f, 0.f, 0.f, 0.f, 0.f, 0.f, 0.f};
    v8f acc1 = {0.f, 0.f, 0.f, 0.f, 0.f, 0.f, 0.f, 0.f};

    const unsigned short* aRow = sH + (rh * 16 + lrow) * LPAD;
    #pragma unroll
    for (int ks = 0; ks < E_DIM / 32; ++ks) {
        const int kb = ks * 32;
        union { v16bf bf; u32x4 q[2]; } A, B0, B1;
        // A fragment: lane holds M=lrow, K = kb + half*8..+8 and kb+16+half*8..+8
        A.q[0] = *reinterpret_cast<const u32x4*>(aRow + kb + half * 8);
        A.q[1] = *reinterpret_cast<const u32x4*>(aRow + kb + 16 + half * 8);
        // B fragment: lane holds N = fb*16+lrow, K = kb + half*8..+8 and kb+16+half*8..+8
        const unsigned short* bp0 = sU + (kb + half * 8) * LPAD + fb0 * 16 + lrow;
        const unsigned short* bp1 = sU + (kb + half * 8) * LPAD + fb1 * 16 + lrow;
        #pragma unroll
        for (int i = 0; i < 4; ++i) {
            B0.q[0][i] = (unsigned)bp0[(2 * i) * LPAD] | ((unsigned)bp0[(2 * i + 1) * LPAD] << 16);
            B0.q[1][i] = (unsigned)bp0[(16 + 2 * i) * LPAD] | ((unsigned)bp0[(17 + 2 * i) * LPAD] << 16);
            B1.q[0][i] = (unsigned)bp1[(2 * i) * LPAD] | ((unsigned)bp1[(2 * i + 1) * LPAD] << 16);
            B1.q[1][i] = (unsigned)bp1[(16 + 2 * i) * LPAD] | ((unsigned)bp1[(17 + 2 * i) * LPAD] << 16);
        }
        acc0 = __builtin_amdgcn_wmma_f32_16x16x32_bf16(
            false, A.bf, false, B0.bf, (short)0, acc0, false, false);
        acc1 = __builtin_amdgcn_wmma_f32_16x16x32_bf16(
            false, A.bf, false, B1.bf, (short)0, acc1, false, false);
    }

    // ---- epilogue: +xw +wv +bias, PReLU, residual, row-sum normalize ----
    const int cf0 = fb0 * 16 + lrow;
    const int cf1 = fb1 * 16 + lrow;
    const float wv0 = wvv[b * E_DIM + cf0];
    const float wv1 = wvv[b * E_DIM + cf1];
    const float bi0 = bias[cf0], bi1 = bias[cf1];
    const float al0 = alpha[cf0], al1 = alpha[cf1];

    float hn0[8], hn1[8];
    #pragma unroll
    for (int j = 0; j < 8; ++j) {
        const int r32 = rh * 16 + half * 8 + j;     // C-layout row mapping
        const int n   = n0 + r32;
        const size_t rowOff = ((size_t)b * N_DIM + n) * E_DIM;
        const float g = gate[r32];
        float p0 = acc0[j] + xw[(size_t)n * E_DIM + cf0] + wv0 + bi0;
        float p1 = acc1[j] + xw[(size_t)n * E_DIM + cf1] + wv1 + bi1;
        p0 = (p0 >= 0.f) ? p0 : al0 * p0;           // PReLU
        p1 = (p1 >= 0.f) ? p1 : al1 * p1;
        const float v0 = h[rowOff + cf0] + g * p0;  // residual (h from L2)
        const float v1 = h[rowOff + cf1] + g * p1;
        hn0[j] = v0; hn1[j] = v1;
        atomicAdd(&rowsum[r32], v0 + v1);           // ds_add_f32
    }
    __syncthreads();
    #pragma unroll
    for (int j = 0; j < 8; ++j) {
        const int r32 = rh * 16 + half * 8 + j;
        const int n   = n0 + r32;
        const size_t rowOff = ((size_t)b * N_DIM + n) * E_DIM;
        const float inv = 1.0f / (rowsum[r32] + 1e-8f);
        out[rowOff + cf0] = hn0[j] * inv;
        out[rowOff + cf1] = hn1[j] * inv;
    }
}

// ---------------- launch ----------------------------------------------------
extern "C" void kernel_launch(void* const* d_in, const int* in_sizes, int n_in,
                              void* d_out, int out_size, void* d_ws, size_t ws_size,
                              hipStream_t stream) {
    (void)in_sizes; (void)n_in; (void)out_size; (void)ws_size;
    const float* x     = (const float*)d_in[0];
    const float* h     = (const float*)d_in[1];
    const float* w_emb = (const float*)d_in[2];
    const float* U     = (const float*)d_in[3];
    const float* V     = (const float*)d_in[4];
    const float* W     = (const float*)d_in[5];
    const float* bias  = (const float*)d_in[6];
    const float* alpha = (const float*)d_in[7];
    float* out = (float*)d_out;

    char* ws = (char*)d_ws;
    unsigned short* Ut = (unsigned short*)ws;                    // 128 KB (bf16 U^T)
    float* xw  = (float*)(ws + 131072);                          // 4 MB
    float* wvv = (float*)(ws + 131072 + 4194304);                // 64 KB
    float* xd  = (float*)(ws + 131072 + 4194304 + 65536);        // 1 MB

    prep_transpose_U<<<E_DIM, E_DIM, 0, stream>>>(U, Ut);
    prep_wv<<<B_DIM, E_DIM, 0, stream>>>(w_emb, V, wvv);
    prep_xw<<<N_DIM, E_DIM, 0, stream>>>(x, W, xw);
    prep_xdot<<<dim3(N_DIM / 256, B_DIM), 256, 0, stream>>>(x, w_emb, xd);

    const int smemBytes = (E_DIM * LPAD + ROWS * LPAD) * 2 + 3 * ROWS * 4; // 152448
    (void)hipFuncSetAttribute((const void*)dm_fused,
                              hipFuncAttributeMaxDynamicSharedMemorySize, smemBytes);
    dm_fused<<<B_DIM * (N_DIM / ROWS), 512, smemBytes, stream>>>(
        x, h, bias, alpha, Ut, xw, wvv, xd, out);
}
